// MFNN0_60017872994948
// MI455X (gfx1250) — matrix operand order repaired and checked by reading, not compiled, not run on hardware
//
#include <hip/hip_runtime.h>

// out[i] = b + W0 + sum_{k=1..128} W_k * cos(k*pi*e[i])
//
// Strategy (compute-bound; see analysis):
//   k = 16q + r (q=0..7, r=1..16);  cos((16q+r)t) = C_q*c_r - S_q*s_r
//   c_r = cos(r*t), s_r = sin(r*t) via Chebyshev recurrence (1 sincos + FMAs)
//   U_q = sum_r W_{16q+r} c_r ; V_q = sum_r W_{16q+r} s_r
//        -> ONE v_wmma_f32_16x16x32_f16 per 16 elements:
//           A (16x32 f16) row m = [c_1..c_16, s_1..s_16] of element m
//           B (32x16 f16) cols 0..7 = cos-weights (K=0..15), cols 8..15 = sin-weights (K=16..31)
//   out_m = bias + sum_q C_q(m)*D[m][q] - S_q(m)*D[m][8+q]
//   D is transposed through per-wave LDS (DS ops are in-order within a wave).

typedef __attribute__((ext_vector_type(16))) _Float16 v16h;
typedef __attribute__((ext_vector_type(8)))  float    v8f;

#define WAVES_PER_BLOCK 8

__global__ __launch_bounds__(256)
void MFNN0_cosfeat_wmma_kernel(const float* __restrict__ e,
                               const float* __restrict__ W,   // 129 floats
                               const float* __restrict__ bp,  // 1 float
                               float* __restrict__ out,
                               int n)
{
    __shared__ float xpose[WAVES_PER_BLOCK][16][16];   // 1 KB per wave

    const int tid    = threadIdx.x;
    const int wave   = tid >> 5;
    const int lane   = tid & 31;
    const int m      = lane & 15;        // element row within 16-wide tile
    const bool hiL   = lane >= 16;       // upper half-lanes supply K=16..31

    // ---------------- B matrix (loop-invariant, wave-uniform weights) -------------
    // Layout (contiguous K order used by this code, same as A): lane n<16 = column N=n with K=0..15 packed
    // 2-per-VGPR; lanes 16..31 = column N=n-16 with K=16..31.
    //   columns 0..7  (U_q): B[K][q]   = W[16q+K+1]  for K=0..15, else 0
    //   columns 8..15 (V_q): B[K][8+q] = W[16q+K-15] for K=16..31, else 0
    v16h Bm;
#pragma unroll
    for (int j = 0; j < 16; ++j) Bm[j] = (_Float16)0.0f;
    if (lane < 8) {                       // cos-weight columns, K=0..15
#pragma unroll
        for (int j = 0; j < 16; ++j) Bm[j] = (_Float16)W[16 * lane + j + 1];
    } else if (lane >= 24) {              // sin-weight columns, K=16..31
#pragma unroll
        for (int j = 0; j < 16; ++j) Bm[j] = (_Float16)W[16 * (lane - 24) + j + 1];
    }
    const float bias = bp[0] + W[0];      // constant feature (cos 0 = 1) + bias

    const int numTiles   = (n + 15) >> 4;
    const int waveId     = blockIdx.x * WAVES_PER_BLOCK + wave;
    const int waveStride = gridDim.x * WAVES_PER_BLOCK;

    for (int tile = waveId; tile < numTiles; tile += waveStride) {
        const int idx = tile * 16 + m;
        const float ev = (idx < n) ? e[idx] : 0.0f;
        const float th = 3.14159265358979323846f * ev;

        float s1, c1;
        __sincosf(th, &s1, &c1);

        // Chebyshev recurrences: c[r]=cos(r*th), s[r]=sin(r*th), r=1..16
        float c[17], s[17];
        c[1] = c1;  s[1] = s1;
        const float twoc = 2.0f * c1;
        c[2] = __builtin_fmaf(twoc, c1, -1.0f);
        s[2] = twoc * s1;
#pragma unroll
        for (int r = 3; r <= 16; ++r) {
            c[r] = __builtin_fmaf(twoc, c[r - 1], -c[r - 2]);
            s[r] = __builtin_fmaf(twoc, s[r - 1], -s[r - 2]);
        }

        // C_q = cos(16q*th), S_q = sin(16q*th) via Chebyshev with step angle 16*th
        float Cq[8], Sq[8];
        Cq[0] = 1.0f;  Sq[0] = 0.0f;
        Cq[1] = c[16]; Sq[1] = s[16];
        const float twoC16 = 2.0f * c[16];
#pragma unroll
        for (int q = 2; q < 8; ++q) {
            Cq[q] = __builtin_fmaf(twoC16, Cq[q - 1], -Cq[q - 2]);
            Sq[q] = __builtin_fmaf(twoC16, Sq[q - 1], -Sq[q - 2]);
        }

        // ------------- A matrix (contiguous K order used by this code, matches B) -------------
        // lane L<16 : row M=L, slots = K 0..15 (c_1..c_16)
        // lane L>=16: row M=L-16, slots = K 16..31 (s_1..s_16)
        v16h A;
        if (!hiL) {
#pragma unroll
            for (int j = 0; j < 16; ++j) {
                A[j]     = (_Float16)c[1 + j];
            }
        } else {
#pragma unroll
            for (int j = 0; j < 16; ++j) {
                A[j]     = (_Float16)s[1 + j];
            }
        }

        // ---------------- WMMA: D[m][q] = U_q(m), D[m][8+q] = V_q(m) ----------------
        v8f D = {};
        D = __builtin_amdgcn_wmma_f32_16x16x32_f16(
                /*neg_a=*/false, A, /*neg_b=*/false, Bm,
                /*c_mod=*/(short)0, D, /*reuse_a=*/false, /*reuse_b=*/false);

        // -------- transpose D through per-wave LDS (DS in-order within a wave) -------
        // D layout: VGPR j, lanes 0-15: row j, col=lane; lanes 16-31: row j+8, col=lane-16
        const int row0 = hiL ? 8 : 0;
#pragma unroll
        for (int j = 0; j < 8; ++j)
            xpose[wave][row0 + j][m] = D[j];

        __builtin_amdgcn_wave_barrier();
        asm volatile("s_wait_dscnt 0x0" ::: "memory");

        // ---------------- per-element combine (lanes 0..15 own element m=lane) -------
        if (!hiL) {
            float acc = bias;
#pragma unroll
            for (int q = 0; q < 8; ++q)
                acc = __builtin_fmaf(Cq[q], xpose[wave][m][q], acc);
#pragma unroll
            for (int q = 0; q < 8; ++q)
                acc = __builtin_fmaf(-Sq[q], xpose[wave][m][8 + q], acc);
            if (idx < n) out[idx] = acc;
        }
        __builtin_amdgcn_wave_barrier();
    }
}

extern "C" void kernel_launch(void* const* d_in, const int* in_sizes, int n_in,
                              void* d_out, int out_size, void* d_ws, size_t ws_size,
                              hipStream_t stream) {
    const float* e  = (const float*)d_in[0];
    const float* W  = (const float*)d_in[1];
    const float* b  = (const float*)d_in[2];
    float* out      = (float*)d_out;
    const int n     = in_sizes[0];

    const int numTiles = (n + 15) / 16;
    int grid = (numTiles + WAVES_PER_BLOCK - 1) / WAVES_PER_BLOCK;
    if (grid > 4096) grid = 4096;           // 32768 waves of grid-stride work
    if (grid < 1) grid = 1;

    MFNN0_cosfeat_wmma_kernel<<<grid, 256, 0, stream>>>(e, W, b, out, n);
}
